// RNNModel_33638183863100
// MI455X (gfx1250) — compile-verified, matrix operand order checked by reading, not run on hardware
//
#include <hip/hip_runtime.h>
#include <math.h>

// ---- Sizes from the reference ----
#define B_   64
#define T_   512
#define DIN_ 512
#define H_   1024
#define OUT_ 128

typedef __bf16 bf16_t;
typedef __attribute__((ext_vector_type(16))) __bf16 v16bf;
typedef __attribute__((ext_vector_type(8)))  __bf16 v8bf;
typedef __attribute__((ext_vector_type(8)))  float  v8f;

// ---------------------------------------------------------------------------
// Load one 16x32 bf16 WMMA fragment (A or B) from a row-major matrix.
// CDNA5 16-bit A-matrix layout (ISA 7.12.2): lane L holds row (L%16);
// elements 0..7  = K in [k0 + kh*8,      k0 + kh*8 + 8)
// elements 8..15 = K in [k0 + 16 + kh*8, k0 + 16 + kh*8 + 8), kh = L/16.
// For D = A*B^T style GEMMs (out[m][n] = sum_k A[m][k]*W[n][k]) the B
// fragment is the identical contiguous-row load from row-major W.
// ---------------------------------------------------------------------------
__device__ __forceinline__ v16bf load_frag(const bf16_t* __restrict__ base,
                                           int ld, int row0, int k0, int lane) {
    const int l16 = lane & 15;
    const int kh  = (lane >> 4) & 1;
    const bf16_t* p = base + (size_t)(row0 + l16) * (size_t)ld + (size_t)(k0 + kh * 8);
    v8bf lo = *reinterpret_cast<const v8bf*>(p);        // 16B load
    v8bf hi = *reinterpret_cast<const v8bf*>(p + 16);   // 16B load
    return __builtin_shufflevector(lo, hi, 0,1,2,3,4,5,6,7,8,9,10,11,12,13,14,15);
}

// ---------------------------------------------------------------------------
// fp32 -> bf16 conversion (grid-stride)
// ---------------------------------------------------------------------------
__global__ void cvt_f32_bf16(const float* __restrict__ in, bf16_t* __restrict__ out,
                             size_t n) {
    size_t i      = (size_t)blockIdx.x * blockDim.x + threadIdx.x;
    size_t stride = (size_t)gridDim.x * blockDim.x;
    for (; i < n; i += stride) out[i] = (bf16_t)in[i];
}

// ---------------------------------------------------------------------------
// pre = A @ W^T + bias_a + bias_b   (bf16 out, f32 accumulate)
// A: MxK row-major bf16, W: NxK row-major bf16.
// Block = 256 threads = 8 waves. Each wave computes a 16x64 slab:
// one A fragment is reused across 4 WMMA tiles; 4 independent accumulator
// chains so WMMAs pipeline against the loads.
// Grid: (N/64, M/128).
// ---------------------------------------------------------------------------
__global__ void wmma_gemm_bias(const bf16_t* __restrict__ A, const bf16_t* __restrict__ W,
                               const float* __restrict__ bias_a, const float* __restrict__ bias_b,
                               bf16_t* __restrict__ out, int M, int N, int K) {
    const int lane = threadIdx.x & 31;
    const int wave = threadIdx.x >> 5;
    const int mt = (blockIdx.y * (blockDim.x >> 5) + wave) * 16;  // 16-row tile start
    const int n0 = blockIdx.x * 64;                               // 64-col slab start
    if (mt >= M) return;

    v8f acc[4] = {v8f{}, v8f{}, v8f{}, v8f{}};
    for (int k0 = 0; k0 < K; k0 += 32) {
        v16bf a  = load_frag(A, K, mt, k0, lane);
        v16bf b0 = load_frag(W, K, n0 +  0, k0, lane);
        v16bf b1 = load_frag(W, K, n0 + 16, k0, lane);
        v16bf b2 = load_frag(W, K, n0 + 32, k0, lane);
        v16bf b3 = load_frag(W, K, n0 + 48, k0, lane);
        acc[0] = __builtin_amdgcn_wmma_f32_16x16x32_bf16(false, a, false, b0, (short)0, acc[0], false, false);
        acc[1] = __builtin_amdgcn_wmma_f32_16x16x32_bf16(false, a, false, b1, (short)0, acc[1], false, false);
        acc[2] = __builtin_amdgcn_wmma_f32_16x16x32_bf16(false, a, false, b2, (short)0, acc[2], false, false);
        acc[3] = __builtin_amdgcn_wmma_f32_16x16x32_bf16(false, a, false, b3, (short)0, acc[3], false, false);
    }

    // C/D layout: VGPR r -> M = r + 8*(lane>=16), N = lane%16
    const int mbase = mt + ((lane >> 4) << 3);
    const int ncol  = lane & 15;
    #pragma unroll
    for (int j = 0; j < 4; ++j) {
        const int n = n0 + j * 16 + ncol;
        const float bv = bias_a[n] + bias_b[n];
        #pragma unroll
        for (int r = 0; r < 8; ++r) {
            out[(size_t)(mbase + r) * (size_t)N + n] = (bf16_t)(acc[j][r] + bv);
        }
    }
}

// ---------------------------------------------------------------------------
// Persistent RNN scan: h_{t} = tanh(pre[:,t,:] + h_{t-1} @ Whh^T)
// Grid: 64 blocks x 128 threads (4 waves). Wave w of block b owns output
// tile rows [16w,16w+16) x cols [16b,16b+16). h double-buffered in global
// scratch; device-wide software barrier (agent atomic + acquire/release
// fences -> global_inv / global_wb) between steps. Whh (2MB bf16) lives in L2.
// K-loop uses 4 independent WMMA accumulator chains (32 WMMAs are on the
// serial critical path every timestep; a single chain would serialize them
// on the D->C RAW hazard).
// ---------------------------------------------------------------------------
__global__ void rnn_scan(const bf16_t* __restrict__ pre,  // (B*T, H) bf16
                         const bf16_t* __restrict__ Whh,  // (H, H) bf16
                         bf16_t* __restrict__ hstate,     // 2 * (B_, H_) bf16, buf0 zeroed
                         bf16_t* __restrict__ hseq,       // (B*T, H) bf16 or null
                         unsigned* __restrict__ bar,      // zeroed
                         int T, int writeSeq) {
    const int lane = threadIdx.x & 31;
    const int wave = threadIdx.x >> 5;       // 0..3
    const int mt   = wave * 16;              // batch-row tile
    const int nt   = blockIdx.x * 16;        // hidden-col tile
    const unsigned nblocks = gridDim.x;

    const int mbase = mt + ((lane >> 4) << 3);
    const int n     = nt + (lane & 15);

    for (int t = 0; t < T; ++t) {
        const bf16_t* hrd = hstate + (size_t)(t & 1) * ((size_t)B_ * H_);
        bf16_t*       hwr = hstate + (size_t)((t + 1) & 1) * ((size_t)B_ * H_);

        v8f acc[4] = {v8f{}, v8f{}, v8f{}, v8f{}};
        #pragma unroll 2
        for (int k0 = 0; k0 < H_; k0 += 128) {
            #pragma unroll
            for (int c = 0; c < 4; ++c) {
                v16bf a = load_frag(hrd, H_, mt, k0 + c * 32, lane);
                v16bf b = load_frag(Whh, H_, nt, k0 + c * 32, lane);
                acc[c] = __builtin_amdgcn_wmma_f32_16x16x32_bf16(false, a, false, b,
                                                                 (short)0, acc[c], false, false);
            }
        }
        v8f accs = (acc[0] + acc[1]) + (acc[2] + acc[3]);

        if (t + 1 < T) {  // hide next step's pre fetch behind the barrier
            __builtin_prefetch(&pre[((size_t)mbase * T + (t + 1)) * H_ + n], 0, 1);
        }

        #pragma unroll
        for (int r = 0; r < 8; ++r) {
            const int brow = mbase + r;  // batch index
            const float p  = (float)pre[((size_t)brow * T + t) * H_ + n];
            const float hn = tanhf(p + accs[r]);
            hwr[(size_t)brow * H_ + n] = (bf16_t)hn;
            if (writeSeq) hseq[((size_t)brow * T + t) * H_ + n] = (bf16_t)hn;
        }

        if (t + 1 < T) {
            // device-wide software barrier between recurrence steps
            __builtin_amdgcn_fence(__ATOMIC_RELEASE, "agent");  // global_wb
            __syncthreads();
            if (threadIdx.x == 0) {
                __hip_atomic_fetch_add(bar, 1u, __ATOMIC_RELAXED, __HIP_MEMORY_SCOPE_AGENT);
                const unsigned target = nblocks * (unsigned)(t + 1);
                while (__hip_atomic_load(bar, __ATOMIC_RELAXED, __HIP_MEMORY_SCOPE_AGENT) < target) {
                    __builtin_amdgcn_s_sleep(2);
                }
            }
            __syncthreads();
            __builtin_amdgcn_fence(__ATOMIC_ACQUIRE, "agent");  // global_inv
        }
    }
}

// ---------------------------------------------------------------------------
// out = h_last @ Wfc^T + b_fc  (fp32 out).  M=64, N=128, K=1024.
// Grid: 8 blocks x 128 threads (4 waves); wave = m-tile, block = n-tile.
// ---------------------------------------------------------------------------
__global__ void wmma_fc(const bf16_t* __restrict__ A,   // (B_, H_) bf16
                        const bf16_t* __restrict__ W,   // (OUT_, H_) bf16
                        const float* __restrict__ bias,
                        float* __restrict__ out) {
    const int lane = threadIdx.x & 31;
    const int wave = threadIdx.x >> 5;
    const int mt = wave * 16;
    const int nt = blockIdx.x * 16;

    v8f acc0 = {}, acc1 = {};
    #pragma unroll 4
    for (int k0 = 0; k0 < H_; k0 += 64) {
        v16bf a0 = load_frag(A, H_, mt, k0, lane);
        v16bf b0 = load_frag(W, H_, nt, k0, lane);
        acc0 = __builtin_amdgcn_wmma_f32_16x16x32_bf16(false, a0, false, b0, (short)0, acc0, false, false);
        v16bf a1 = load_frag(A, H_, mt, k0 + 32, lane);
        v16bf b1 = load_frag(W, H_, nt, k0 + 32, lane);
        acc1 = __builtin_amdgcn_wmma_f32_16x16x32_bf16(false, a1, false, b1, (short)0, acc1, false, false);
    }
    v8f acc = acc0 + acc1;

    const int mbase = mt + ((lane >> 4) << 3);
    const int n     = nt + (lane & 15);
    const float bv  = bias[n];
    #pragma unroll
    for (int r = 0; r < 8; ++r) {
        out[(size_t)(mbase + r) * OUT_ + n] = acc[r] + bv;
    }
}

// ---------------------------------------------------------------------------
extern "C" void kernel_launch(void* const* d_in, const int* in_sizes, int n_in,
                              void* d_out, int out_size, void* d_ws, size_t ws_size,
                              hipStream_t stream) {
    (void)in_sizes; (void)n_in; (void)out_size; (void)ws_size;

    const float* x    = (const float*)d_in[0];
    const float* Wih0 = (const float*)d_in[1];
    const float* Whh0 = (const float*)d_in[2];
    const float* bih0 = (const float*)d_in[3];
    const float* bhh0 = (const float*)d_in[4];
    const float* Wih1 = (const float*)d_in[5];
    const float* Whh1 = (const float*)d_in[6];
    const float* bih1 = (const float*)d_in[7];
    const float* bhh1 = (const float*)d_in[8];
    const float* Wfc  = (const float*)d_in[9];
    const float* bfc  = (const float*)d_in[10];
    float* out = (float*)d_out;

    // ---- scratch layout (bf16 staging buffers), 256B aligned ----
    char* ws = (char*)d_ws;
    size_t off = 0;
    auto alloc = [&](size_t bytes) {
        char* p = ws + off;
        off = (off + bytes + 255) & ~(size_t)255;
        return p;
    };
    const size_t MT = (size_t)B_ * T_;
    bf16_t* xb     = (bf16_t*)alloc(MT * DIN_ * sizeof(bf16_t));       // 32 MB
    bf16_t* h1b    = (bf16_t*)alloc(MT * H_   * sizeof(bf16_t));       // 64 MB
    bf16_t* preb   = (bf16_t*)alloc(MT * H_   * sizeof(bf16_t));       // 64 MB
    bf16_t* wih0b  = (bf16_t*)alloc((size_t)H_ * DIN_ * sizeof(bf16_t));
    bf16_t* whh0b  = (bf16_t*)alloc((size_t)H_ * H_   * sizeof(bf16_t));
    bf16_t* wih1b  = (bf16_t*)alloc((size_t)H_ * H_   * sizeof(bf16_t));
    bf16_t* whh1b  = (bf16_t*)alloc((size_t)H_ * H_   * sizeof(bf16_t));
    bf16_t* wfcb   = (bf16_t*)alloc((size_t)OUT_ * H_ * sizeof(bf16_t));
    bf16_t* hstate = (bf16_t*)alloc(2 * (size_t)B_ * H_ * sizeof(bf16_t)); // double buffer
    unsigned* bar  = (unsigned*)alloc(256);

    // ---- 1) fp32 -> bf16 staging ----
    dim3 cb(256), cg(1024);
    cvt_f32_bf16<<<cg, cb, 0, stream>>>(x,    xb,    MT * DIN_);
    cvt_f32_bf16<<<cg, cb, 0, stream>>>(Wih0, wih0b, (size_t)H_ * DIN_);
    cvt_f32_bf16<<<cg, cb, 0, stream>>>(Whh0, whh0b, (size_t)H_ * H_);
    cvt_f32_bf16<<<cg, cb, 0, stream>>>(Wih1, wih1b, (size_t)H_ * H_);
    cvt_f32_bf16<<<cg, cb, 0, stream>>>(Whh1, whh1b, (size_t)H_ * H_);
    cvt_f32_bf16<<<cg, cb, 0, stream>>>(Wfc,  wfcb,  (size_t)OUT_ * H_);

    // ---- 2) pre0 = x @ Wih0^T + bih0 + bhh0 ----
    {
        dim3 grid(H_ / 64, (int)(MT / 128));
        wmma_gemm_bias<<<grid, 256, 0, stream>>>(xb, wih0b, bih0, bhh0, preb,
                                                 (int)MT, H_, DIN_);
    }

    // ---- 3) layer-0 recurrence ----
    hipMemsetAsync(hstate, 0, (size_t)B_ * H_ * sizeof(bf16_t), stream); // buf0 = h_{-1} = 0
    hipMemsetAsync(bar, 0, sizeof(unsigned), stream);
    rnn_scan<<<dim3(H_ / 16), 128, 0, stream>>>(preb, whh0b, hstate, h1b, bar, T_, 1);

    // ---- 4) pre1 = h1 @ Wih1^T + bih1 + bhh1 ----
    {
        dim3 grid(H_ / 64, (int)(MT / 128));
        wmma_gemm_bias<<<grid, 256, 0, stream>>>(h1b, wih1b, bih1, bhh1, preb,
                                                 (int)MT, H_, H_);
    }

    // ---- 5) layer-1 recurrence (only final h needed) ----
    hipMemsetAsync(hstate, 0, (size_t)B_ * H_ * sizeof(bf16_t), stream);
    hipMemsetAsync(bar, 0, sizeof(unsigned), stream);
    rnn_scan<<<dim3(H_ / 16), 128, 0, stream>>>(preb, whh1b, hstate, nullptr, bar, T_, 0);

    // T_ is even -> final h lives in buffer 0 of hstate
    // ---- 6) out = h_last @ Wfc^T + b_fc ----
    wmma_fc<<<dim3(OUT_ / 16), 128, 0, stream>>>(hstate, wfcb, bfc, out);
}
